// WaveInputEncoder_81853486727583
// MI455X (gfx1250) — matrix operand order repaired and checked by reading, not compiled, not run on hardware
//
#include <hip/hip_runtime.h>
#include <math.h>

#define B_   4
#define T_   4096
#define D_   1024
#define M_   32
#define NB_  8
#define BS_  128
#define HH_  4096
#define NM_  32
#define EPS_ 1e-5f
#define LAM_ 0.01f
#define INV_SQRT_T (1.0f/64.0f)   // ortho norm, sqrt(4096)=64

typedef __attribute__((ext_vector_type(16))) __bf16 v16bf;
typedef __attribute__((ext_vector_type(8)))  __bf16 v8bf;
typedef __attribute__((ext_vector_type(8)))  float  v8f;
typedef __attribute__((ext_vector_type(4)))  int    v4i;

// ---- gfx1250 async global->LDS path (guarded so both toolchains compile) ----
#if defined(__has_builtin)
# if __has_builtin(__builtin_amdgcn_global_load_async_to_lds_b128)
#  define HAVE_ASYNC_LDS 1
# endif
# if __has_builtin(__builtin_amdgcn_s_wait_asynccnt)
#  define HAVE_WAIT_ASYNC 1
# endif
#endif

typedef __attribute__((address_space(1))) v4i glb_v4i;
typedef __attribute__((address_space(3))) v4i lds_v4i;

__device__ __forceinline__ void async_copy16(const __bf16* g, __bf16* s) {
#if defined(HAVE_ASYNC_LDS)
    __builtin_amdgcn_global_load_async_to_lds_b128(
        (glb_v4i*)(uintptr_t)g, (lds_v4i*)(uintptr_t)s, 0, 0);
#else
    *(v8bf*)s = *(const v8bf*)g;
#endif
}

__device__ __forceinline__ void wait_async_copies() {
#if defined(HAVE_ASYNC_LDS)
# if defined(HAVE_WAIT_ASYNC)
    __builtin_amdgcn_s_wait_asynccnt(0);
# else
    asm volatile("s_wait_asynccnt 0x0" ::: "memory");
# endif
#endif
}

// ---------------------------------------------------------------- utilities

__global__ void zero_kernel(float* p) {
    p[(long)blockIdx.x * 256 + threadIdx.x] = 0.f;
}

// twiddle tables: twc[k*T+t] = cos(2*pi*k*t/T), tws = sin(...)
__global__ __launch_bounds__(256) void twiddle_kernel(float* twc, float* tws) {
    int idx = blockIdx.x * 256 + threadIdx.x;     // idx = k*T_ + t
    int k = idx >> 12;
    int t = idx & (T_ - 1);
    double th = (2.0 * M_PI / (double)T_) * (double)(k * t);
    twc[idx] = (float)cos(th);
    tws[idx] = (float)sin(th);
}

// mlp_w1 (D,H) fp32 row-major -> w1t[h*D + d] bf16 (transposed)
__global__ __launch_bounds__(256) void convert_w1_kernel(const float* w, __bf16* wt) {
    long idx = (long)blockIdx.x * 256 + threadIdx.x;   // over D_*HH_
    int d = (int)(idx >> 12);          // / HH_
    int h = (int)(idx & (HH_ - 1));
    wt[(long)h * D_ + d] = (__bf16)w[idx];
}

// mlp_w2 (H,D) fp32 row-major -> w2t[d*H + h] bf16 (transposed)
__global__ __launch_bounds__(256) void convert_w2_kernel(const float* w, __bf16* wt) {
    long idx = (long)blockIdx.x * 256 + threadIdx.x;   // over HH_*D_
    int h = (int)(idx >> 10);          // / D_
    int d = (int)(idx & (D_ - 1));
    wt[(long)d * HH_ + h] = (__bf16)w[idx];
}

// ------------------------------------------------- stage 1: token -> x_res
__global__ __launch_bounds__(256) void encode_kernel(
    const int* __restrict__ tokens, const float* __restrict__ frw,
    const float* __restrict__ fiw, const float* __restrict__ mw,
    const float* __restrict__ ph, const float* __restrict__ fte_w,
    const float* __restrict__ fte_b, const float* __restrict__ pos,
    float* __restrict__ x_res) {
    __shared__ float ff[2 * M_];
    __shared__ float xrow[D_];
    int bt  = blockIdx.x;
    int t   = bt & (T_ - 1);
    int tid = threadIdx.x;
    int tok = tokens[bt];
    if (tid < M_) {
        float r  = frw[(long)tok * M_ + tid];
        float im = fiw[(long)tok * M_ + tid];
        float w  = mw[tid];
        w = (w > 20.f) ? w : log1pf(expf(w));       // softplus
        r *= w; im *= w;
        float c = cosf(ph[tid]), s = sinf(ph[tid]);
        ff[tid]      = r * c - im * s;
        ff[M_ + tid] = r * s + im * c;
    }
    __syncthreads();
    #pragma unroll
    for (int u = 0; u < 4; ++u) {
        int d = tid + u * 256;
        float acc = fte_b[d];
        for (int q = 0; q < 2 * M_; ++q)
            acc += ff[q] * fte_w[q * D_ + d];
        xrow[d] = acc;
    }
    __syncthreads();
    #pragma unroll
    for (int u = 0; u < 4; ++u) {
        int d = tid + u * 256;
        float pp = pos[(long)t * D_ + d];
        float v = xrow[d] * cosf(pp) + xrow[(d + D_ - 1) & (D_ - 1)] * sinf(pp);
        x_res[(long)bt * D_ + d] = v;
    }
}

// ----------------------------------------- stage 2: truncated forward DFT
__global__ __launch_bounds__(256) void dft_fwd_kernel(
    const float* __restrict__ x_res, const float* __restrict__ twc,
    const float* __restrict__ tws, float* __restrict__ xr, float* __restrict__ xi) {
    __shared__ float sc[NM_ * 64];
    __shared__ float ss[NM_ * 64];
    int bn  = blockIdx.x;            // b*8 + n
    int b   = bn >> 3;
    int n   = bn & 7;
    int tid = threadIdx.x;
    int j   = tid & (BS_ - 1);
    int kh  = tid >> 7;              // 0/1 -> 16 k's each
    float ar[16], ai[16];
    #pragma unroll
    for (int q = 0; q < 16; ++q) { ar[q] = 0.f; ai[q] = 0.f; }
    int tbeg = blockIdx.y * (T_ / 8);
    for (int t0 = tbeg; t0 < tbeg + T_ / 8; t0 += 64) {
        __syncthreads();
        #pragma unroll
        for (int q = 0; q < 8; ++q) {
            int idx = tid * 8 + q;             // k*64 + tt
            int k  = idx >> 6;
            int tt = idx & 63;
            sc[idx] = twc[k * T_ + t0 + tt];
            ss[idx] = tws[k * T_ + t0 + tt];
        }
        __syncthreads();
        for (int tt = 0; tt < 64; ++tt) {
            float xv = x_res[(long)(b * T_ + t0 + tt) * D_ + n * BS_ + j];
            int kb = kh * 16;
            #pragma unroll
            for (int kk = 0; kk < 16; ++kk) {
                float c = sc[(kb + kk) * 64 + tt];
                float s = ss[(kb + kk) * 64 + tt];
                ar[kk] += xv * c;
                ai[kk] -= xv * s;
            }
        }
    }
    #pragma unroll
    for (int kk = 0; kk < 16; ++kk) {
        int k = kh * 16 + kk;
        long o = ((long)bn * NM_ + k) * BS_ + j;
        atomicAdd(&xr[o], ar[kk] * INV_SQRT_T);
        atomicAdd(&xi[o], ai[kk] * INV_SQRT_T);
    }
}

// ------------------------------------ stage 3: per-mode complex 128x128 mv
__global__ __launch_bounds__(128) void afno_mode_kernel(
    const float* __restrict__ wr, const float* __restrict__ wi,
    const float* __restrict__ xr, const float* __restrict__ xi,
    float* __restrict__ orr, float* __restrict__ oii) {
    __shared__ float sxr[BS_], sxi[BS_];
    int bid = blockIdx.x;            // b*256 + n*32 + m
    int b  = bid >> 8;
    int nm = bid & 255;
    int n  = nm >> 5;
    int m  = nm & 31;
    int i  = threadIdx.x;
    long xoff = (((long)(b * NB_ + n)) * NM_ + m) * BS_;
    sxr[i] = xr[xoff + i];
    sxi[i] = xi[xoff + i];
    __syncthreads();
    long woff = (((long)(n * M_ + m)) * BS_ + i) * BS_;
    float accr = 0.f, acci = 0.f;
    for (int jv = 0; jv < BS_; ++jv) {
        float a  = wr[woff + jv];
        float bb = wi[woff + jv];
        accr += a * sxr[jv] - bb * sxi[jv];
        acci += a * sxi[jv] + bb * sxr[jv];
    }
    accr = (accr > LAM_) ? accr - LAM_ : ((accr < -LAM_) ? accr + LAM_ : 0.f);
    acci = (acci > LAM_) ? acci - LAM_ : ((acci < -LAM_) ? acci + LAM_ : 0.f);
    orr[xoff + i] = accr;
    oii[xoff + i] = acci;
}

// ------------------- stage 4: 32-mode irfft + residual + LN1 + bf16 cast
__global__ __launch_bounds__(256) void idft_res_ln1_kernel(
    const float* __restrict__ orr, const float* __restrict__ oii,
    const float* __restrict__ twc, const float* __restrict__ tws,
    const float* __restrict__ x_res, const float* __restrict__ g,
    const float* __restrict__ bvec, float* __restrict__ y1,
    __bf16* __restrict__ y1bf) {
    __shared__ float sc[NM_], ss[NM_];
    __shared__ float red[256];
    int bt  = blockIdx.x;
    int b   = bt >> 12;
    int t   = bt & (T_ - 1);
    int tid = threadIdx.x;
    if (tid < NM_) { sc[tid] = twc[tid * T_ + t]; ss[tid] = tws[tid * T_ + t]; }
    __syncthreads();
    float vals[4];
    float psum = 0.f;
    #pragma unroll
    for (int u = 0; u < 4; ++u) {
        int d = tid + u * 256;
        int n = d >> 7;
        int j = d & 127;
        long base = ((long)(b * NB_ + n)) * NM_ * BS_ + j;
        float acc = orr[base] * sc[0] - oii[base] * ss[0];
        for (int k = 1; k < NM_; ++k)
            acc += 2.f * (orr[base + (long)k * BS_] * sc[k]
                        - oii[base + (long)k * BS_] * ss[k]);
        float v = acc * INV_SQRT_T + x_res[(long)bt * D_ + d];
        vals[u] = v;
        psum += v;
    }
    red[tid] = psum; __syncthreads();
    for (int sft = 128; sft > 0; sft >>= 1) {
        if (tid < sft) red[tid] += red[tid + sft];
        __syncthreads();
    }
    float mu = red[0] * (1.f / D_);
    __syncthreads();
    float pvar = 0.f;
    #pragma unroll
    for (int u = 0; u < 4; ++u) { float dd = vals[u] - mu; pvar += dd * dd; }
    red[tid] = pvar; __syncthreads();
    for (int sft = 128; sft > 0; sft >>= 1) {
        if (tid < sft) red[tid] += red[tid + sft];
        __syncthreads();
    }
    float inv = rsqrtf(red[0] * (1.f / D_) + EPS_);
    #pragma unroll
    for (int u = 0; u < 4; ++u) {
        int d = tid + u * 256;
        float yv = (vals[u] - mu) * inv * g[d] + bvec[d];
        y1[(long)bt * D_ + d]   = yv;
        y1bf[(long)bt * D_ + d] = (__bf16)yv;
    }
}

// ----------------------------------------------------------- WMMA GEMMs
// Block: 8 waves, 32 M-rows shared via LDS (async double-buffered),
// each wave owns a 32x64 C tile: 2 A-frags x 4 B-frags -> 8 WMMA / K-step.
#define AP_ 40   // LDS row pitch in bf16 elems (80B: bank-conflict-free, 16B aligned)

// stage 5: h = gelu(y1 @ W1 + b1)   M=16384 K=1024 N=4096
__global__ __launch_bounds__(256) void mlp_gemm1_kernel(
    const __bf16* __restrict__ A, const __bf16* __restrict__ Bt,
    const float* __restrict__ bias, __bf16* __restrict__ Hout) {
    const int K = D_, N = HH_;
    __shared__ __bf16 sA[2][32 * AP_];
    int tid  = threadIdx.x;
    int wid  = tid >> 5;
    int lane = tid & 31;
    int m0 = blockIdx.x * 32;
    int n0 = (blockIdx.y * 8 + wid) * 64;
    int col = lane & 15;
    int hi  = lane >> 4;
    v8f vzero = {0.f, 0.f, 0.f, 0.f, 0.f, 0.f, 0.f, 0.f};
    v8f c[2][4];
    #pragma unroll
    for (int i = 0; i < 2; ++i)
        #pragma unroll
        for (int jx = 0; jx < 4; ++jx) c[i][jx] = vzero;

    auto stage = [&](int buf, int kk) {
        if (tid < 128) {
            int r = tid >> 2, seg = tid & 3;
            async_copy16(A + (long)(m0 + r) * K + kk + seg * 8,
                         &sA[buf][r * AP_ + seg * 8]);
        }
    };
    stage(0, 0);
    int buf = 0;
    for (int kk = 0; kk < K; kk += 32) {
        wait_async_copies();
        __syncthreads();
        if (kk + 32 < K) stage(buf ^ 1, kk + 32);
        v16bf a[2];
        #pragma unroll
        for (int mt = 0; mt < 2; ++mt) {
            const __bf16* arow = &sA[buf][(mt * 16 + col) * AP_];
            v8bf alo = *(const v8bf*)(arow + hi * 8);
            v8bf ahi = *(const v8bf*)(arow + 16 + hi * 8);
            a[mt] = __builtin_shufflevector(alo, ahi,
                0, 1, 2, 3, 4, 5, 6, 7, 8, 9, 10, 11, 12, 13, 14, 15);
        }
        #pragma unroll
        for (int nt = 0; nt < 4; ++nt) {
            const __bf16* bp = Bt + (long)(n0 + nt * 16 + col) * K + kk;
            v8bf blo = *(const v8bf*)(bp + hi * 8);
            v8bf bhi = *(const v8bf*)(bp + 16 + hi * 8);
            v16bf bfrag = __builtin_shufflevector(blo, bhi,
                0, 1, 2, 3, 4, 5, 6, 7, 8, 9, 10, 11, 12, 13, 14, 15);
            c[0][nt] = __builtin_amdgcn_wmma_f32_16x16x32_bf16(
                false, a[0], false, bfrag, (short)0, c[0][nt], false, false);
            c[1][nt] = __builtin_amdgcn_wmma_f32_16x16x32_bf16(
                false, a[1], false, bfrag, (short)0, c[1][nt], false, false);
        }
        __syncthreads();
        buf ^= 1;
    }
    #pragma unroll
    for (int mt = 0; mt < 2; ++mt)
        #pragma unroll
        for (int nt = 0; nt < 4; ++nt)
            #pragma unroll
            for (int e = 0; e < 8; ++e) {
                int m = m0 + mt * 16 + hi * 8 + e;
                int n = n0 + nt * 16 + col;
                float v = c[mt][nt][e] + bias[n];
                v = 0.5f * v * (1.0f + erff(v * 0.70710678118f));
                Hout[(long)m * N + n] = (__bf16)v;
            }
}

// stage 6: out = y1 + h @ W2 + b2   M=16384 K=4096 N=1024
__global__ __launch_bounds__(256) void mlp_gemm2_kernel(
    const __bf16* __restrict__ A, const __bf16* __restrict__ Bt,
    const float* __restrict__ bias, const float* __restrict__ res,
    float* __restrict__ out) {
    const int K = HH_, N = D_;
    __shared__ __bf16 sA[2][32 * AP_];
    int tid  = threadIdx.x;
    int wid  = tid >> 5;
    int lane = tid & 31;
    int m0 = blockIdx.x * 32;
    int n0 = (blockIdx.y * 8 + wid) * 64;
    int col = lane & 15;
    int hi  = lane >> 4;
    v8f vzero = {0.f, 0.f, 0.f, 0.f, 0.f, 0.f, 0.f, 0.f};
    v8f c[2][4];
    #pragma unroll
    for (int i = 0; i < 2; ++i)
        #pragma unroll
        for (int jx = 0; jx < 4; ++jx) c[i][jx] = vzero;

    auto stage = [&](int buf, int kk) {
        if (tid < 128) {
            int r = tid >> 2, seg = tid & 3;
            async_copy16(A + (long)(m0 + r) * K + kk + seg * 8,
                         &sA[buf][r * AP_ + seg * 8]);
        }
    };
    stage(0, 0);
    int buf = 0;
    for (int kk = 0; kk < K; kk += 32) {
        wait_async_copies();
        __syncthreads();
        if (kk + 32 < K) stage(buf ^ 1, kk + 32);
        v16bf a[2];
        #pragma unroll
        for (int mt = 0; mt < 2; ++mt) {
            const __bf16* arow = &sA[buf][(mt * 16 + col) * AP_];
            v8bf alo = *(const v8bf*)(arow + hi * 8);
            v8bf ahi = *(const v8bf*)(arow + 16 + hi * 8);
            a[mt] = __builtin_shufflevector(alo, ahi,
                0, 1, 2, 3, 4, 5, 6, 7, 8, 9, 10, 11, 12, 13, 14, 15);
        }
        #pragma unroll
        for (int nt = 0; nt < 4; ++nt) {
            const __bf16* bp = Bt + (long)(n0 + nt * 16 + col) * K + kk;
            v8bf blo = *(const v8bf*)(bp + hi * 8);
            v8bf bhi = *(const v8bf*)(bp + 16 + hi * 8);
            v16bf bfrag = __builtin_shufflevector(blo, bhi,
                0, 1, 2, 3, 4, 5, 6, 7, 8, 9, 10, 11, 12, 13, 14, 15);
            c[0][nt] = __builtin_amdgcn_wmma_f32_16x16x32_bf16(
                false, a[0], false, bfrag, (short)0, c[0][nt], false, false);
            c[1][nt] = __builtin_amdgcn_wmma_f32_16x16x32_bf16(
                false, a[1], false, bfrag, (short)0, c[1][nt], false, false);
        }
        __syncthreads();
        buf ^= 1;
    }
    #pragma unroll
    for (int mt = 0; mt < 2; ++mt)
        #pragma unroll
        for (int nt = 0; nt < 4; ++nt)
            #pragma unroll
            for (int e = 0; e < 8; ++e) {
                int m = m0 + mt * 16 + hi * 8 + e;
                int n = n0 + nt * 16 + col;
                out[(long)m * N + n] = c[mt][nt][e] + bias[n] + res[(long)m * N + n];
            }
}

// ------------------------------------------- stage 7: LN2 in-place on d_out
__global__ __launch_bounds__(256) void ln2_kernel(
    float* __restrict__ x, const float* __restrict__ g,
    const float* __restrict__ bvec) {
    __shared__ float red[256];
    long base = (long)blockIdx.x * D_;
    int tid = threadIdx.x;
    float vals[4];
    float psum = 0.f;
    #pragma unroll
    for (int u = 0; u < 4; ++u) {
        vals[u] = x[base + tid + u * 256];
        psum += vals[u];
    }
    red[tid] = psum; __syncthreads();
    for (int sft = 128; sft > 0; sft >>= 1) {
        if (tid < sft) red[tid] += red[tid + sft];
        __syncthreads();
    }
    float mu = red[0] * (1.f / D_);
    __syncthreads();
    float pvar = 0.f;
    #pragma unroll
    for (int u = 0; u < 4; ++u) { float dd = vals[u] - mu; pvar += dd * dd; }
    red[tid] = pvar; __syncthreads();
    for (int sft = 128; sft > 0; sft >>= 1) {
        if (tid < sft) red[tid] += red[tid + sft];
        __syncthreads();
    }
    float inv = rsqrtf(red[0] * (1.f / D_) + EPS_);
    #pragma unroll
    for (int u = 0; u < 4; ++u) {
        int d = tid + u * 256;
        x[base + d] = (vals[u] - mu) * inv * g[d] + bvec[d];
    }
}

// ---------------------------------------------------------------- launcher

extern "C" void kernel_launch(void* const* d_in, const int* in_sizes, int n_in,
                              void* d_out, int out_size, void* d_ws, size_t ws_size,
                              hipStream_t stream) {
    const int*   tokens = (const int*)  d_in[0];
    const float* frw    = (const float*)d_in[1];
    const float* fiw    = (const float*)d_in[2];
    const float* mw     = (const float*)d_in[3];
    const float* ph     = (const float*)d_in[4];
    const float* fte_w  = (const float*)d_in[5];
    const float* fte_b  = (const float*)d_in[6];
    const float* pos    = (const float*)d_in[7];
    const float* awr    = (const float*)d_in[8];
    const float* awi    = (const float*)d_in[9];
    const float* ln1g   = (const float*)d_in[10];
    const float* ln1b   = (const float*)d_in[11];
    const float* w1     = (const float*)d_in[12];
    const float* b1     = (const float*)d_in[13];
    const float* w2     = (const float*)d_in[14];
    const float* b2     = (const float*)d_in[15];
    const float* ln2g   = (const float*)d_in[16];
    const float* ln2b   = (const float*)d_in[17];

    char* ws = (char*)d_ws;
    float*  x_res = (float*) (ws + 0L);                 //  67108864 B
    float*  y1    = (float*) (ws + 67108864L);          //  67108864 B
    __bf16* y1bf  = (__bf16*)(ws + 134217728L);         //  33554432 B
    __bf16* hbf   = (__bf16*)(ws + 167772160L);         // 134217728 B
    __bf16* w1t   = (__bf16*)(ws + 301989888L);         //   8388608 B
    __bf16* w2t   = (__bf16*)(ws + 310378496L);         //   8388608 B
    float*  twc   = (float*) (ws + 318767104L);         //    524288 B
    float*  tws   = (float*) (ws + 319291392L);         //    524288 B
    float*  xr    = (float*) (ws + 319815680L);         // 4 x 524288 B
    const int SPEC = B_ * NB_ * NM_ * BS_;              // 131072 elems
    float*  xi    = xr + SPEC;
    float*  orr   = xi + SPEC;
    float*  oii   = orr + SPEC;
    float*  outf  = (float*)d_out;

    twiddle_kernel<<<(NM_ * T_) / 256, 256, 0, stream>>>(twc, tws);
    convert_w1_kernel<<<(D_ * HH_) / 256, 256, 0, stream>>>(w1, w1t);
    convert_w2_kernel<<<(HH_ * D_) / 256, 256, 0, stream>>>(w2, w2t);
    zero_kernel<<<(2 * SPEC) / 256, 256, 0, stream>>>(xr);   // xr and xi

    encode_kernel<<<B_ * T_, 256, 0, stream>>>(
        tokens, frw, fiw, mw, ph, fte_w, fte_b, pos, x_res);

    dft_fwd_kernel<<<dim3(B_ * NB_, 8), 256, 0, stream>>>(x_res, twc, tws, xr, xi);

    afno_mode_kernel<<<B_ * NB_ * NM_, 128, 0, stream>>>(awr, awi, xr, xi, orr, oii);

    idft_res_ln1_kernel<<<B_ * T_, 256, 0, stream>>>(
        orr, oii, twc, tws, x_res, ln1g, ln1b, y1, y1bf);

    mlp_gemm1_kernel<<<dim3((B_ * T_) / 32, HH_ / 512), 256, 0, stream>>>(
        y1bf, w1t, b1, hbf);

    mlp_gemm2_kernel<<<dim3((B_ * T_) / 32, D_ / 512), 256, 0, stream>>>(
        hbf, w2t, b2, y1, outf);

    ln2_kernel<<<B_ * T_, 256, 0, stream>>>(outf, ln2g, ln2b);
}
